// nodeselection_10161892622585
// MI455X (gfx1250) — matrix operand order repaired and checked by reading, not compiled, not run on hardware
//
#include <hip/hip_runtime.h>

// ---------------------------------------------------------------------------
// NodeSelection for MI455X (gfx1250, wave32, WMMA)
//
// scores = emb[M,D] x feat[b,N,D]^T  -> top-10 indices per (b,m) -> gather.
// softmax is monotonic and its values are unused, so it is skipped entirely.
//
// Bandwidth-bound: 82 MB feature read / 23.3 TB/s ~ 3.5 us floor. GEMM done
// with v_wmma_f32_16x16x32_bf16 using a hi/lo bf16 split of the fp32 inputs
// (3 WMMAs per k-step) for ~fp32-accurate scores at negligible cost.
// A-operand (embeddings tile) lives in VGPRs for the whole kernel.
// ---------------------------------------------------------------------------

typedef __attribute__((ext_vector_type(16))) __bf16 v16bf;
typedef __attribute__((ext_vector_type(8)))  float  v8f;

constexpr int Bn  = 32;      // batch
constexpr int Nn  = 10000;   // nodes
constexpr int Dn  = 64;      // feature dim (= WMMA K total, 2 k-steps of 32)
constexpr int Mn  = 64;      // embeddings
constexpr int KT  = 10;      // top-k
constexpr int CH  = 512;     // n-chunk handled per block (32 tiles of 16)
constexpr int NCH = (Nn + CH - 1) / CH;  // 20 chunks

#define NEGINF (-3.0e38f)

// ---------------------------------------------------------------------------
// Kernel 1: one block = (batch b, m-tile of 16, n-chunk of 512).
// 4 waves: WMMA-score the 16x512 tile into LDS, then per-row wave top-10.
// ---------------------------------------------------------------------------
__global__ __launch_bounds__(128, 1)
void nodesel_score_topk(const float* __restrict__ feat,
                        const float* __restrict__ emb,
                        float* __restrict__ cand_val,
                        int*   __restrict__ cand_idx)
{
    __shared__ float sc[16 * CH];   // 32 KB score tile [16 m][512 n]

    const int b    = blockIdx.x;
    const int mt   = blockIdx.y;
    const int ch   = blockIdx.z;
    const int lane = threadIdx.x & 31;
    const int wave = threadIdx.x >> 5;
    const int h    = lane >> 4;     // half-wave select
    const int l15  = lane & 15;

    // ---- A operand: embeddings rows mt*16..mt*16+15, fp32 -> bf16 hi+lo ----
    // ISA 16-bit A 16x32 layout: lane(l<16)=row M, element j maps to
    // K = 8h + (j<8 ? j : j+8)  (i.e. {8h..8h+7} U {16+8h..16+8h+7}).
    v16bf ahi[2], alo[2];
    {
        const float* arow = emb + (mt * 16 + l15) * Dn;
        #pragma unroll
        for (int s = 0; s < 2; ++s) {
            const int kb = s * 32 + h * 8;
            const float4 q0 = *(const float4*)(arow + kb);
            const float4 q1 = *(const float4*)(arow + kb + 4);
            const float4 q2 = *(const float4*)(arow + kb + 16);
            const float4 q3 = *(const float4*)(arow + kb + 20);
            const float xs[16] = {q0.x,q0.y,q0.z,q0.w, q1.x,q1.y,q1.z,q1.w,
                                  q2.x,q2.y,q2.z,q2.w, q3.x,q3.y,q3.z,q3.w};
            #pragma unroll
            for (int j = 0; j < 16; ++j) {
                const __bf16 hi = (__bf16)xs[j];
                ahi[s][j] = hi;
                alo[s][j] = (__bf16)(xs[j] - (float)hi);
            }
        }
    }

    // ---- Phase A: stream feature tiles, WMMA, scores -> LDS ----
    const float* fb = feat + (size_t)b * Nn * Dn;

    for (int t = wave; t < CH / 16; t += 4) {           // uniform per wave
        const int ng = ch * CH + t * 16 + l15;          // global n (column)
        const int nc = (ng < Nn) ? ng : (Nn - 1);       // clamp; masked below
        const float* frow = fb + (size_t)nc * Dn;

        // ISA 16-bit B 32x16 layout: lane = column, K = 8h + (j<8 ? j : j+8) (matches A).
        v16bf bhi[2], blo[2];
        #pragma unroll
        for (int s = 0; s < 2; ++s) {
            const int kb = s * 32 + h * 8;
            const float4 q0 = *(const float4*)(frow + kb);
            const float4 q1 = *(const float4*)(frow + kb + 4);
            const float4 q2 = *(const float4*)(frow + kb + 16);
            const float4 q3 = *(const float4*)(frow + kb + 20);
            const float xs[16] = {q0.x,q0.y,q0.z,q0.w, q1.x,q1.y,q1.z,q1.w,
                                  q2.x,q2.y,q2.z,q2.w, q3.x,q3.y,q3.z,q3.w};
            #pragma unroll
            for (int j = 0; j < 16; ++j) {
                const __bf16 hi = (__bf16)xs[j];
                bhi[s][j] = hi;
                blo[s][j] = (__bf16)(xs[j] - (float)hi);
            }
        }

        // D = A*B + C with hi/lo split: hh + hl + lh (lo*lo dropped).
        v8f c = {};
        #pragma unroll
        for (int s = 0; s < 2; ++s) {
            c = __builtin_amdgcn_wmma_f32_16x16x32_bf16(
                    false, ahi[s], false, bhi[s], (short)0, c, false, false);
            c = __builtin_amdgcn_wmma_f32_16x16x32_bf16(
                    false, ahi[s], false, blo[s], (short)0, c, false, false);
            c = __builtin_amdgcn_wmma_f32_16x16x32_bf16(
                    false, alo[s], false, bhi[s], (short)0, c, false, false);
        }

        // C/D layout: element v -> row v+8h, column lane&15.
        #pragma unroll
        for (int v = 0; v < 8; ++v) {
            const int row = v + 8 * h;
            sc[row * CH + t * 16 + l15] = (ng < Nn) ? c[v] : NEGINF;
        }
    }
    __syncthreads();

    // ---- Phase B: per-row top-10 by successive wave-max (tie: smaller idx) ----
    volatile float* vsc = sc;
    for (int row = wave; row < 16; row += 4) {
        const int m = mt * 16 + row;
        for (int k = 0; k < KT; ++k) {
            float best = NEGINF;
            int   bcol = 0x7fffffff;
            for (int cix = lane; cix < CH; cix += 32) {
                const float x = vsc[row * CH + cix];
                if (x > best || (x == best && cix < bcol)) { best = x; bcol = cix; }
            }
            #pragma unroll
            for (int msk = 16; msk >= 1; msk >>= 1) {
                const float ov = __shfl_xor(best, msk, 32);
                const int   oc = __shfl_xor(bcol, msk, 32);
                if (ov > best || (ov == best && oc < bcol)) { best = ov; bcol = oc; }
            }
            if (lane == 0) {
                const size_t o = (((size_t)b * Mn + m) * NCH + ch) * KT + k;
                cand_val[o] = best;
                cand_idx[o] = ch * CH + bcol;
                vsc[row * CH + bcol] = NEGINF;     // remove winner (same wave)
            }
        }
    }
}

// ---------------------------------------------------------------------------
// Kernel 2: one block per (b,m). Merge 200 chunk-candidates -> final top-10
// (sorted desc, ties by smaller index), then gather D=64 features per pick.
// ---------------------------------------------------------------------------
__global__ __launch_bounds__(64, 1)
void nodesel_merge_gather(const float* __restrict__ feat,
                          const float* __restrict__ cand_val,
                          const int*   __restrict__ cand_idx,
                          float* __restrict__ out_sel,
                          float* __restrict__ out_bidx,
                          float* __restrict__ out_idx)
{
    const int bm  = blockIdx.x;           // b*Mn + m
    const int b   = bm / Mn;
    const int tid = threadIdx.x;

    __shared__ float vals[NCH * KT];
    __shared__ int   idxs[NCH * KT];
    __shared__ int   sel[KT];

    for (int i = tid; i < NCH * KT; i += 64) {
        vals[i] = cand_val[(size_t)bm * NCH * KT + i];
        idxs[i] = cand_idx[(size_t)bm * NCH * KT + i];
    }
    __syncthreads();

    if (tid == 0) {                       // 200 candidates: trivial serial pick
        for (int k = 0; k < KT; ++k) {
            float best = NEGINF; int bi = 0x7fffffff; int bp = 0;
            for (int i = 0; i < NCH * KT; ++i) {
                const float v = vals[i]; const int ix = idxs[i];
                if (v > best || (v == best && ix < bi)) { best = v; bi = ix; bp = i; }
            }
            sel[k]   = bi;
            vals[bp] = NEGINF;
        }
    }
    __syncthreads();

    const float* fbase = feat + (size_t)b * Nn * Dn;
    for (int k = 0; k < KT; ++k) {        // 64 threads gather one row each pass
        const int nidx = sel[k];
        out_sel[((size_t)bm * KT + k) * Dn + tid] = fbase[(size_t)nidx * Dn + tid];
    }
    if (tid < KT) {
        out_bidx[(size_t)bm * KT + tid] = (float)b;
        out_idx [(size_t)bm * KT + tid] = (float)sel[tid];
    }
}

// ---------------------------------------------------------------------------
extern "C" void kernel_launch(void* const* d_in, const int* in_sizes, int n_in,
                              void* d_out, int out_size, void* d_ws, size_t ws_size,
                              hipStream_t stream)
{
    const float* feat = (const float*)d_in[0];   // [B,N,D]
    const float* emb  = (const float*)d_in[1];   // [M,D]

    // workspace: per-chunk candidate lists (values then indices), ~3.3 MB
    float* cand_val = (float*)d_ws;
    int*   cand_idx = (int*)((char*)d_ws +
                     (size_t)Bn * Mn * NCH * KT * sizeof(float));

    // outputs concatenated flat: selected [B,M,K,D] f32, batch_indices [B,M,K],
    // indices [B,M,K] (written as float values into the float* output buffer)
    float* out_sel  = (float*)d_out;
    float* out_bidx = out_sel  + (size_t)Bn * Mn * KT * Dn;
    float* out_idx  = out_bidx + (size_t)Bn * Mn * KT;

    dim3 g1(Bn, Mn / 16, NCH);
    nodesel_score_topk<<<g1, 128, 0, stream>>>(feat, emb, cand_val, cand_idx);

    nodesel_merge_gather<<<Bn * Mn, 64, 0, stream>>>(
        feat, cand_val, cand_idx, out_sel, out_bidx, out_idx);
}
